// MolecularGnnSmiles_44014824849805
// MI455X (gfx1250) — matrix operand order repaired and checked by reading, not compiled, not run on hardware
//
#include <hip/hip_runtime.h>

#define HID 128

typedef __attribute__((ext_vector_type(16))) __bf16          v16bf;
typedef __attribute__((ext_vector_type(16))) unsigned short  v16us;
typedef __attribute__((ext_vector_type(8)))  unsigned short  v8us;
typedef __attribute__((ext_vector_type(8)))  float           v8f;
typedef __attribute__((ext_vector_type(4)))  unsigned int    v4u;
typedef __attribute__((ext_vector_type(8)))  int             v8i;
typedef __attribute__((ext_vector_type(4)))  int             v4i;

union BF16x16 { v16us u; v16bf b; v8us h[2]; };

__device__ __forceinline__ unsigned short f2bf(float f) {
    __bf16 h = (__bf16)f;                       // native v_cvt path
    return __builtin_bit_cast(unsigned short, h);
}
__device__ __forceinline__ unsigned int pack2bf(float a, float b) {
    return (unsigned)f2bf(a) | ((unsigned)f2bf(b) << 16);
}

// ---------------------------------------------------------------- zero (float4)
__global__ __launch_bounds__(256) void zero4_kernel(float4* __restrict__ p, int n4) {
    int i = blockIdx.x * blockDim.x + threadIdx.x;
    if (i < n4) p[i] = make_float4(0.f, 0.f, 0.f, 0.f);
}

// ------------------------------------------------- weights fp32 -> bf16 bits
__global__ __launch_bounds__(256) void convert_w_kernel(
        const float* __restrict__ gcn_w, const float* __restrict__ lin_w,
        unsigned short* __restrict__ WB, int n_gcn, int n_lin) {
    int i = blockIdx.x * blockDim.x + threadIdx.x;
    if (i < n_gcn)              WB[i] = f2bf(gcn_w[i]);
    else if (i < n_gcn + n_lin) WB[i] = f2bf(lin_w[i - n_gcn]);
}

// ------------------------------------------ embedding: write X (f32) + XB (bf16)
__global__ __launch_bounds__(256) void embed_kernel(
        const int* __restrict__ ids, const float* __restrict__ embd,
        float* __restrict__ X, unsigned short* __restrict__ XB, int nrows) {
    const int lane = threadIdx.x & 31;
    const int i = blockIdx.x * 8 + (threadIdx.x >> 5);
    if (i >= nrows) return;
    const int id = ids[i];
    const float4 v = *(const float4*)(embd + (size_t)id * HID + lane * 4);
    *(float4*)(X + (size_t)i * HID + lane * 4) = v;
    uint2 pk; pk.x = pack2bf(v.x, v.y); pk.y = pack2bf(v.z, v.w);
    *(uint2*)(XB + (size_t)i * HID + lane * 4) = pk;
}

// ------------------------------------------ h = relu(x @ W^T + b), h in bf16
// TDM stages the shared 128x128 bf16 weight tile into LDS once per block;
// 8 waves then feed WMMA B operands from LDS. A loads stay bf16 b128 global.
__global__ __launch_bounds__(256) void linear_relu_wmma_kernel(
        const unsigned short* __restrict__ XB, const unsigned short* __restrict__ WB,
        const float* __restrict__ bias, unsigned short* __restrict__ H, int nrows) {
    __shared__ unsigned short wlds[HID * HID];       // 32KB: weight tile (TDM dest)
    __shared__ unsigned short tile[8][16 * HID];     // 32KB: per-wave output staging
    const int lane = threadIdx.x & 31;
    const int wid  = threadIdx.x >> 5;

    // ---- stage weights into LDS
#if __has_builtin(__builtin_amdgcn_tensor_load_to_lds)
    if (wid == 0) {
        const unsigned lds_off = (unsigned)(size_t)(void*)&wlds[0];
        const unsigned long long ga = (unsigned long long)(size_t)WB;
        v4u g0;
        g0[0] = 1u;                                       // count=1 user D#
        g0[1] = lds_off;                                  // lds_addr (bytes)
        g0[2] = (unsigned)(ga & 0xffffffffu);             // global_addr[31:0]
        g0[3] = (unsigned)((ga >> 32) & 0x01ffffffu)      // global_addr[56:32]
              | (2u << 30);                               // type=2 ("image")
        v8i g1;
        g1[0] = (int)0x00010000u;   // workgroup_mask=0, data_size=1 (2 bytes)
        g1[1] = (int)(128u << 16);  // tensor_dim0[15:0]=128 @ bit48
        g1[2] = (int)(128u << 16);  // tensor_dim0 hi=0, tensor_dim1[15:0]=128 @ bit80
        g1[3] = (int)(128u << 16);  // tensor_dim1 hi=0, tile_dim0=128 @ bit112
        g1[4] = (int)128u;          // tile_dim1=128, tile_dim2=0
        g1[5] = (int)128u;          // tensor_dim0_stride[31:0]=128
        g1[6] = (int)0x40000000u;   // stride hi=0, tensor_dim1_stride[15:0]=16384
        g1[7] = 0;                  // tensor_dim1_stride hi
        v4i gz = {0, 0, 0, 0};      // 2D tensor: groups 2/3 unused
#if defined(__clang_major__) && (__clang_major__ >= 23)
        v8i gz8 = {0, 0, 0, 0, 0, 0, 0, 0};
        __builtin_amdgcn_tensor_load_to_lds(g0, g1, gz, gz, gz8, 0);
#else
        __builtin_amdgcn_tensor_load_to_lds(g0, g1, gz, gz, 0);
#endif
#if __has_builtin(__builtin_amdgcn_s_wait_tensorcnt)
        __builtin_amdgcn_s_wait_tensorcnt(0);
#else
        asm volatile("s_wait_tensorcnt 0x0" ::: "memory");
#endif
    }
#else
    // fallback: cooperative vector copy, 2048 x 16B chunks over 256 threads
    {
        const uint4* gsrc = (const uint4*)WB;
        uint4* ldst = (uint4*)wlds;
#pragma unroll
        for (int i = 0; i < 8; ++i)
            ldst[threadIdx.x + 256 * i] = gsrc[threadIdx.x + 256 * i];
    }
#endif
    __syncthreads();

    const int mtile = blockIdx.x * 8 + wid;
    const int m0 = mtile * 16;
    if (m0 >= nrows) return;
    const int half = lane >> 4;
    const int nl   = lane & 15;
    unsigned short* tw = tile[wid];

    // A operand in ISA 16-bit layout: elems 0..7 = K kc*32+half*8+[0,8),
    // elems 8..15 = K kc*32+16+half*8+[0,8) -> two b128 loads per chunk.
    const unsigned short* xr = XB + (size_t)(m0 + nl) * HID;
    BF16x16 a[4];
#pragma unroll
    for (int kc = 0; kc < 4; ++kc) {
        a[kc].h[0] = *(const v8us*)(xr + kc * 32 + half * 8);
        a[kc].h[1] = *(const v8us*)(xr + kc * 32 + 16 + half * 8);
    }

#pragma unroll
    for (int nt = 0; nt < 8; ++nt) {
        const int n0 = nt * 16;
        v8f c = {};
#pragma unroll
        for (int kc = 0; kc < 4; ++kc) {
            BF16x16 b;   // B from LDS: lane nl = column n0+nl, K = kc*32+half*8+[0,8), kc*32+16+half*8+[0,8)
            b.h[0] = *(const v8us*)(wlds + (n0 + nl) * HID + kc * 32 + half * 8);
            b.h[1] = *(const v8us*)(wlds + (n0 + nl) * HID + kc * 32 + 16 + half * 8);
            c = __builtin_amdgcn_wmma_f32_16x16x32_bf16(
                    false, a[kc].b, false, b.b, (short)0, c, false, false);
        }
        const float bv = bias[n0 + nl];
#pragma unroll
        for (int j = 0; j < 8; ++j) {
            float v = c[j] + bv;
            v = v > 0.f ? v : 0.f;
            tw[(half * 8 + j) * HID + n0 + nl] = f2bf(v);   // DS scatter (cheap)
        }
    }

    // Wave-private LDS (DScnt is in-order) -> 8 coalesced b128 stores per lane.
    uint4* __restrict__ dstp = (uint4*)(H + (size_t)m0 * HID);
#pragma unroll
    for (int i = 0; i < 8; ++i) {
        const int chunk = i * 32 + lane;                    // 256 x 16B chunks
        uint4 q = *(const uint4*)(tw + chunk * 8);
        dstp[chunk] = q;
    }
}

// ---------------------------------------- agg[dst] += h[src]  (wave per edge)
__global__ __launch_bounds__(256) void edge_scatter_kernel(
        const int* __restrict__ src, const int* __restrict__ dst,
        const unsigned short* __restrict__ H, float* __restrict__ AGG, int E) {
    const int lane = threadIdx.x & 31;
    const int e = blockIdx.x * 8 + (threadIdx.x >> 5);
    if (e >= E) return;
    const int s = src[e];
    const int d = dst[e];
    const unsigned short* hr = H + (size_t)s * HID;
    __builtin_prefetch(hr, 0, 3);                      // global_prefetch_b8
    const uint2 p = *(const uint2*)(hr + lane * 4);    // 4 bf16 per lane
    const float f0 = __uint_as_float(p.x << 16);
    const float f1 = __uint_as_float(p.x & 0xffff0000u);
    const float f2 = __uint_as_float(p.y << 16);
    const float f3 = __uint_as_float(p.y & 0xffff0000u);
    float* o = AGG + (size_t)d * HID + lane * 4;
    __hip_atomic_fetch_add(o + 0, f0, __ATOMIC_RELAXED, __HIP_MEMORY_SCOPE_AGENT);
    __hip_atomic_fetch_add(o + 1, f1, __ATOMIC_RELAXED, __HIP_MEMORY_SCOPE_AGENT);
    __hip_atomic_fetch_add(o + 2, f2, __ATOMIC_RELAXED, __HIP_MEMORY_SCOPE_AGENT);
    __hip_atomic_fetch_add(o + 3, f3, __ATOMIC_RELAXED, __HIP_MEMORY_SCOPE_AGENT);
}

// ------------- x = (agg + x)/max(||agg+x||,eps); also refresh bf16 mirror XB
__global__ __launch_bounds__(256) void residual_norm_kernel(
        const float* __restrict__ AGG, float* __restrict__ X,
        unsigned short* __restrict__ XB, int nrows) {
    const int lane = threadIdx.x & 31;
    const int i = blockIdx.x * 8 + (threadIdx.x >> 5);
    if (i >= nrows) return;
    const float4 ag = *(const float4*)(AGG + (size_t)i * HID + lane * 4);
    float* xp = X + (size_t)i * HID + lane * 4;
    const float4 xv = *(const float4*)xp;
    const float r0 = ag.x + xv.x, r1 = ag.y + xv.y;
    const float r2 = ag.z + xv.z, r3 = ag.w + xv.w;
    float ss = r0 * r0 + r1 * r1 + r2 * r2 + r3 * r3;
#pragma unroll
    for (int off = 16; off >= 1; off >>= 1) ss += __shfl_xor(ss, off, 32);
    const float scale = 1.f / fmaxf(sqrtf(ss), 1e-12f);
    float4 o; o.x = r0 * scale; o.y = r1 * scale; o.z = r2 * scale; o.w = r3 * scale;
    *(float4*)xp = o;
    uint2 pk; pk.x = pack2bf(o.x, o.y); pk.y = pack2bf(o.z, o.w);
    *(uint2*)(XB + (size_t)i * HID + lane * 4) = pk;
}

// ------------------------------------------------- m[batch[i]] += x[i]
__global__ __launch_bounds__(256) void pool_kernel(
        const float* __restrict__ X, const int* __restrict__ batch,
        float* __restrict__ M, int nrows) {
    const int lane = threadIdx.x & 31;
    const int i = blockIdx.x * 8 + (threadIdx.x >> 5);
    if (i >= nrows) return;
    const int g = batch[i];
    const float4 v = *(const float4*)(X + (size_t)i * HID + lane * 4);
    float* o = M + (size_t)g * HID + lane * 4;
    __hip_atomic_fetch_add(o + 0, v.x, __ATOMIC_RELAXED, __HIP_MEMORY_SCOPE_AGENT);
    __hip_atomic_fetch_add(o + 1, v.y, __ATOMIC_RELAXED, __HIP_MEMORY_SCOPE_AGENT);
    __hip_atomic_fetch_add(o + 2, v.z, __ATOMIC_RELAXED, __HIP_MEMORY_SCOPE_AGENT);
    __hip_atomic_fetch_add(o + 3, v.w, __ATOMIC_RELAXED, __HIP_MEMORY_SCOPE_AGENT);
}

// ------------------- fused head: relu(relu(m W0^T+b0) W1^T+b1) . wp + wpb
// One wave (32 threads) per block, one 16-row tile per block.
__global__ __launch_bounds__(32) void mlp_head_kernel(
        const float* __restrict__ M,
        const unsigned short* __restrict__ W0, const unsigned short* __restrict__ W1,
        const float* __restrict__ B0, const float* __restrict__ B1,
        const float* __restrict__ wpw, const float* __restrict__ wpb,
        float* __restrict__ out, int ngraphs) {
    __shared__ unsigned short lds[16 * HID];
    const int lane = threadIdx.x & 31;
    const int half = lane >> 4;
    const int nl   = lane & 15;
    const int m0 = blockIdx.x * 16;
    if (m0 >= ngraphs) return;

    // ---- layer 1: A from pooled fp32
    BF16x16 a[4];
    const float* xr = M + (size_t)(m0 + nl) * HID;
#pragma unroll
    for (int kc = 0; kc < 4; ++kc) {
        const float* p0 = xr + kc * 32 + half * 8;
        const float* p1 = xr + kc * 32 + 16 + half * 8;
#pragma unroll
        for (int j = 0; j < 8; ++j) {
            a[kc].u[j]     = f2bf(p0[j]);
            a[kc].u[8 + j] = f2bf(p1[j]);
        }
    }
#pragma unroll
    for (int nt = 0; nt < 8; ++nt) {
        const int n0 = nt * 16;
        v8f c = {};
#pragma unroll
        for (int kc = 0; kc < 4; ++kc) {
            BF16x16 b;
            b.h[0] = *(const v8us*)(W0 + (size_t)(n0 + nl) * HID + kc * 32 + half * 8);
            b.h[1] = *(const v8us*)(W0 + (size_t)(n0 + nl) * HID + kc * 32 + 16 + half * 8);
            c = __builtin_amdgcn_wmma_f32_16x16x32_bf16(
                    false, a[kc].b, false, b.b, (short)0, c, false, false);
        }
        const float bv = B0[n0 + nl];
#pragma unroll
        for (int j = 0; j < 8; ++j) {
            float v = c[j] + bv;
            v = v > 0.f ? v : 0.f;
            lds[(half * 8 + j) * HID + n0 + nl] = f2bf(v);
        }
    }
    __syncthreads();

    // ---- layer 2: A reloaded from LDS in A-operand layout
    BF16x16 a2[4];
    const unsigned short* lr = lds + (size_t)nl * HID;
#pragma unroll
    for (int kc = 0; kc < 4; ++kc) {
        a2[kc].h[0] = *(const v8us*)(lr + kc * 32 + half * 8);
        a2[kc].h[1] = *(const v8us*)(lr + kc * 32 + 16 + half * 8);
    }
    float acc[8];
#pragma unroll
    for (int j = 0; j < 8; ++j) acc[j] = 0.f;
#pragma unroll
    for (int nt = 0; nt < 8; ++nt) {
        const int n0 = nt * 16;
        v8f c = {};
#pragma unroll
        for (int kc = 0; kc < 4; ++kc) {
            BF16x16 b;
            b.h[0] = *(const v8us*)(W1 + (size_t)(n0 + nl) * HID + kc * 32 + half * 8);
            b.h[1] = *(const v8us*)(W1 + (size_t)(n0 + nl) * HID + kc * 32 + 16 + half * 8);
            c = __builtin_amdgcn_wmma_f32_16x16x32_bf16(
                    false, a2[kc].b, false, b.b, (short)0, c, false, false);
        }
        const float bv = B1[n0 + nl];
        const float wv = wpw[n0 + nl];
#pragma unroll
        for (int j = 0; j < 8; ++j) {
            float v = c[j] + bv;
            v = v > 0.f ? v : 0.f;
            acc[j] += v * wv;          // final projection folded in
        }
    }
    // reduce across the 16 lanes that share each output row
#pragma unroll
    for (int j = 0; j < 8; ++j) {
#pragma unroll
        for (int off = 8; off >= 1; off >>= 1)
            acc[j] += __shfl_xor(acc[j], off, 16);
    }
    const float wb = wpb[0];
    if (nl == 0) {                      // lanes 0 (rows 0..7) and 16 (rows 8..15)
#pragma unroll
        for (int j = 0; j < 8; ++j)
            out[m0 + half * 8 + j] = acc[j] + wb;
    }
}

// ============================================================================
extern "C" void kernel_launch(void* const* d_in, const int* in_sizes, int n_in,
                              void* d_out, int out_size, void* d_ws, size_t ws_size,
                              hipStream_t stream) {
    const int*   x_ids = (const int*)d_in[0];
    const int*   edges = (const int*)d_in[1];
    const int*   batch = (const int*)d_in[2];
    const float* embd  = (const float*)d_in[3];
    const float* gcn_w = (const float*)d_in[4];
    const float* gcn_b = (const float*)d_in[5];
    const float* lin_w = (const float*)d_in[6];
    const float* lin_b = (const float*)d_in[7];
    const float* wp_w  = (const float*)d_in[8];
    const float* wp_b  = (const float*)d_in[9];
    float*       out   = (float*)d_out;

    const int N = in_sizes[0];          // 50000 nodes
    const int E = in_sizes[1] / 2;      // 600000 edges
    const int G = out_size;             // 1024 graphs
    const int* src = edges;
    const int* dst = edges + E;

    // ---- workspace carve-up (256B-aligned offsets)
    char* ws = (char*)d_ws;
    float*          X    = (float*)(ws);                             // N*128 f32
    float*          AGG  = (float*)(ws + (size_t)25600000);          // N*128 f32
    unsigned short* H    = (unsigned short*)(ws + (size_t)51200000); // N*128 bf16
    unsigned short* XB   = (unsigned short*)(ws + (size_t)64000000); // N*128 bf16
    unsigned short* WB   = (unsigned short*)(ws + (size_t)76800000); // 5*128*128 bf16
    float*          MBUF = (float*)(ws + (size_t)76963840);          // G*128 f32
    (void)ws_size;

    const int n_gcn = 3 * HID * HID, n_lin = 2 * HID * HID;
    const int blkN  = (N + 7) / 8;                 // wave-per-node kernels
    const int blkM  = ((N + 15) / 16 + 7) / 8;     // wave-per-16-row-tile GEMM
    const int blkE  = (E + 7) / 8;                 // wave-per-edge scatter
    const int blkG  = (G + 15) / 16;               // head tiles

    convert_w_kernel<<<(n_gcn + n_lin + 255) / 256, 256, 0, stream>>>(
        gcn_w, lin_w, WB, n_gcn, n_lin);
    embed_kernel<<<blkN, 256, 0, stream>>>(x_ids, embd, X, XB, N);

    for (int l = 0; l < 3; ++l) {
        linear_relu_wmma_kernel<<<blkM, 256, 0, stream>>>(
            XB, WB + (size_t)l * HID * HID, gcn_b + (size_t)l * HID, H, N);
        zero4_kernel<<<(N * HID / 4 + 255) / 256, 256, 0, stream>>>(
            (float4*)AGG, N * HID / 4);
        edge_scatter_kernel<<<blkE, 256, 0, stream>>>(src, dst, H, AGG, E);
        residual_norm_kernel<<<blkN, 256, 0, stream>>>(AGG, X, XB, N);
    }

    zero4_kernel<<<(G * HID / 4 + 255) / 256, 256, 0, stream>>>(
        (float4*)MBUF, G * HID / 4);
    pool_kernel<<<blkN, 256, 0, stream>>>(X, batch, MBUF, N);

    mlp_head_kernel<<<blkG, 32, 0, stream>>>(
        MBUF, WB + (size_t)3 * HID * HID, WB + (size_t)4 * HID * HID,
        lin_b, lin_b + HID, wp_w, wp_b, out, G);
}
